// MultiHeadAttention_87462714015835
// MI455X (gfx1250) — matrix operand order repaired and checked by reading, not compiled, not run on hardware
//
#include <hip/hip_runtime.h>

#define BATCH  4
#define SEQ    2048
#define DMODEL 1024
#define NHEAD  8
#define DK     128
#define LN_EPS 1e-5f
#define KC     64   // attention key-chunk
#define TKQ    64   // qkv staged K-slice (2 WMMA k-steps per barrier)

typedef __attribute__((ext_vector_type(16))) _Float16 v16h;
typedef __attribute__((ext_vector_type(8)))  _Float16 v8h;
typedef __attribute__((ext_vector_type(4)))  _Float16 v4h;
typedef __attribute__((ext_vector_type(8)))  float    v8f;

// D = A(16x32 f16) * B(32x16 f16) + C(16x16 f32), wave32
__device__ __forceinline__ v8f wmma16(v16h a, v16h b, v8f c) {
  return __builtin_amdgcn_wmma_f32_16x16x32_f16(false, a, false, b, (short)0, c,
                                                false, false);
}

// Assemble a v16h fragment from two aligned 16-byte pieces.
__device__ __forceinline__ v16h ld_frag(const _Float16* lo, const _Float16* hi) {
  v8h l = *(const v8h*)lo;
  v8h h = *(const v8h*)hi;
  v16h r;
#pragma unroll
  for (int i = 0; i < 8; ++i) { r[i] = l[i]; r[i + 8] = h[i]; }
  return r;
}

// ---------------------------------------------------------------------------
// Kernel 1: Y = (X @ W^T + bias) * outScale, f32 in -> f16 out.
// vtrans==0: Y layout [B,H,S,dk] ; vtrans==1: Y layout [B,H,dk,S] (for V).
// Grid: (M/128, NHEAD), block 256 (8 waves). Tile 128x128, staged K-slice 64.
// ---------------------------------------------------------------------------
__global__ __launch_bounds__(256)
void qkv_proj_kernel(const float* __restrict__ X, const float* __restrict__ W,
                     const float* __restrict__ bias, _Float16* __restrict__ Y,
                     int vtrans, float outScale) {
  __shared__ _Float16 As[128][TKQ + 8];  // [m][k], +8 pad (stride 144B)
  __shared__ _Float16 Bs[128][TKQ + 8];  // [n][k] == W rows

  const int tid  = threadIdx.x;
  const int wv   = tid >> 5;
  const int lane = tid & 31;
  const int nl   = lane & 15;
  const int msel = lane >> 4;
  const int mtile = blockIdx.x;  // 64 tiles of 128 rows
  const int h     = blockIdx.y;  // head == N-tile of 128

  v8f acc[8] = {};

  const int kbA = msel ? 8 : 0;   // A frag K base within 32-chunk
  const int kbB = msel ? 8 : 0;   // B frag K base within 32-chunk
  const int ma  = wv * 16 + nl;   // A row within tile

  for (int k0 = 0; k0 < DMODEL; k0 += TKQ) {
    __syncthreads();
#pragma unroll
    for (int i = 0; i < 8; ++i) {  // 2048 float4 per array
      int idx = tid + i * 256;
      int row = idx >> 4;          // 16 float4 per 64-wide row
      int c4  = idx & 15;
      float4 xa = *(const float4*)&X[(size_t)(mtile * 128 + row) * DMODEL + k0 + c4 * 4];
      float4 xb = *(const float4*)&W[(size_t)(h * 128 + row) * DMODEL + k0 + c4 * 4];
      v4h ha = {(_Float16)xa.x, (_Float16)xa.y, (_Float16)xa.z, (_Float16)xa.w};
      v4h hb = {(_Float16)xb.x, (_Float16)xb.y, (_Float16)xb.z, (_Float16)xb.w};
      *(v4h*)&As[row][c4 * 4] = ha;
      *(v4h*)&Bs[row][c4 * 4] = hb;
    }
    __syncthreads();

    // prefetch next K-slice while WMMAs run (global_prefetch_b8)
    if (k0 + TKQ < DMODEL) {
      int idx = tid, row = idx >> 4, c4 = idx & 15;
      __builtin_prefetch(&X[(size_t)(mtile * 128 + row) * DMODEL + k0 + TKQ + c4 * 4], 0, 1);
      __builtin_prefetch(&W[(size_t)(h * 128 + row) * DMODEL + k0 + TKQ + c4 * 4], 0, 1);
      int idx2 = tid + 1024, row2 = idx2 >> 4, c42 = idx2 & 15;
      __builtin_prefetch(&X[(size_t)(mtile * 128 + row2) * DMODEL + k0 + TKQ + c42 * 4], 0, 1);
      __builtin_prefetch(&W[(size_t)(h * 128 + row2) * DMODEL + k0 + TKQ + c42 * 4], 0, 1);
    }

#pragma unroll
    for (int kk = 0; kk < TKQ; kk += 32) {
      // Hoist: batch all fragment loads, then run WMMAs back-to-back.
      v16h af = ld_frag(&As[ma][kk + kbA], &As[ma][kk + kbA + 16]);
      v16h bfr[8];
#pragma unroll
      for (int j = 0; j < 8; ++j)
        bfr[j] = ld_frag(&Bs[j * 16 + nl][kk + kbB], &Bs[j * 16 + nl][kk + kbB + 16]);
#pragma unroll
      for (int j = 0; j < 8; ++j) acc[j] = wmma16(af, bfr[j], acc[j]);
    }
  }

  const int gm0  = mtile * 128 + wv * 16 + msel * 8;  // first row this lane holds
  const int bidx = (mtile * 128) >> 11;               // batch (SEQ=2048)
  if (!vtrans) {
#pragma unroll
    for (int j = 0; j < 8; ++j) {
      int dd = j * 16 + nl;
      float bb = bias[h * DK + dd];
#pragma unroll
      for (int v = 0; v < 8; ++v) {
        int s = (gm0 + v) & (SEQ - 1);
        Y[(((size_t)bidx * NHEAD + h) * SEQ + s) * DK + dd] =
            (_Float16)((acc[j][v] + bb) * outScale);
      }
    }
  } else {
    int s0 = gm0 & (SEQ - 1);
#pragma unroll
    for (int j = 0; j < 8; ++j) {
      int dd = j * 16 + nl;
      float bb = bias[h * DK + dd];
      v8h pk;
#pragma unroll
      for (int v = 0; v < 8; ++v) pk[v] = (_Float16)((acc[j][v] + bb) * outScale);
      *(v8h*)&Y[(((size_t)bidx * NHEAD + h) * DK + dd) * SEQ + s0] = pk;  // V^T
    }
  }
}

// ---------------------------------------------------------------------------
// Kernel 2: flash attention. Q pre-scaled by 1/sqrt(dk).
// Q,K: [B,H,S,dk] f16 ; Vt: [B,H,dk,S] f16 ; O: [B,H,S,dk] f32.
// Grid: (S/128, B*H), block 256 (8 waves, 16 queries/wave).
// ---------------------------------------------------------------------------
__global__ __launch_bounds__(256)
void attention_kernel(const _Float16* __restrict__ Q, const _Float16* __restrict__ K,
                      const _Float16* __restrict__ Vt, float* __restrict__ O) {
  __shared__ _Float16 Ks[KC][DK + 8];      // [key][d]
  __shared__ _Float16 Vs[DK][KC + 8];      // [d][key]  (V^T slice)
  __shared__ _Float16 Ps[8][16][KC + 8];   // per-wave P transpose scratch

  const int tid  = threadIdx.x;
  const int wv   = tid >> 5;
  const int lane = tid & 31;
  const int nl   = lane & 15;
  const int msel = lane >> 4;
  const int qtile = blockIdx.x;
  const int bh    = blockIdx.y;
  const size_t base = (size_t)bh * SEQ * DK;

  const int kbA = msel ? 8 : 0;
  const int kbB = msel ? 8 : 0;

  // Resident Q fragments: 4 K-steps over dk=128
  const int qm = qtile * 128 + wv * 16 + nl;
  v16h aq[4];
#pragma unroll
  for (int t = 0; t < 4; ++t)
    aq[t] = ld_frag(&Q[base + (size_t)qm * DK + t * 32 + kbA],
                    &Q[base + (size_t)qm * DK + t * 32 + kbA + 16]);

  v8f o[8] = {};
  float mrow[8], lrow[8];
#pragma unroll
  for (int v = 0; v < 8; ++v) { mrow[v] = -1e30f; lrow[v] = 0.0f; }

  for (int kc = 0; kc < SEQ; kc += KC) {
    __syncthreads();
    // stage K chunk: 64 keys x 128 d  (1024 v8h)
#pragma unroll
    for (int i = 0; i < 4; ++i) {
      int idx = tid + i * 256;
      int row = idx >> 4;  // 16 v8h per key row
      int c   = idx & 15;
      *(v8h*)&Ks[row][c * 8] = *(const v8h*)&K[base + (size_t)(kc + row) * DK + c * 8];
    }
    // stage V^T chunk: 128 d x 64 keys (1024 v8h)
#pragma unroll
    for (int i = 0; i < 4; ++i) {
      int idx = tid + i * 256;
      int row = idx >> 3;  // 8 v8h per d row
      int c   = idx & 7;
      *(v8h*)&Vs[row][c * 8] = *(const v8h*)&Vt[base + (size_t)row * SEQ + kc + c * 8];
    }
    __syncthreads();

    // prefetch next key chunk while this one computes
    if (kc + KC < SEQ) {
      int idx = tid, row = idx >> 4, c = idx & 15;
      __builtin_prefetch(&K[base + (size_t)(kc + KC + row) * DK + c * 8], 0, 1);
      int row2 = tid >> 3, c2 = tid & 7;
      __builtin_prefetch(&Vt[base + (size_t)row2 * SEQ + kc + KC + c2 * 8], 0, 1);
    }

    // scores: 16q x 64keys ; hoist B-frags per k-step, then 4 WMMAs back-to-back
    v8f sc[4] = {};
#pragma unroll
    for (int t = 0; t < 4; ++t) {
      v16h bfr[4];
#pragma unroll
      for (int j = 0; j < 4; ++j)
        bfr[j] = ld_frag(&Ks[j * 16 + nl][t * 32 + kbB],
                         &Ks[j * 16 + nl][t * 32 + kbB + 16]);
#pragma unroll
      for (int j = 0; j < 4; ++j) sc[j] = wmma16(aq[t], bfr[j], sc[j]);
    }

    // online softmax (row = v + msel*8, 16-lane groups hold one row's keys)
    float p[4][8];
#pragma unroll
    for (int v = 0; v < 8; ++v) {
      float mx = sc[0][v];
#pragma unroll
      for (int j = 1; j < 4; ++j) mx = fmaxf(mx, sc[j][v]);
#pragma unroll
      for (int m = 8; m >= 1; m >>= 1) mx = fmaxf(mx, __shfl_xor(mx, m));
      float mnew  = fmaxf(mrow[v], mx);
      float alpha = __expf(mrow[v] - mnew);
      float rs = 0.0f;
#pragma unroll
      for (int j = 0; j < 4; ++j) {
        p[j][v] = __expf(sc[j][v] - mnew);
        rs += p[j][v];
      }
#pragma unroll
      for (int m = 8; m >= 1; m >>= 1) rs += __shfl_xor(rs, m);
      lrow[v] = lrow[v] * alpha + rs;
      mrow[v] = mnew;
#pragma unroll
      for (int jj = 0; jj < 8; ++jj) o[jj][v] *= alpha;
    }

    // transpose P via per-wave LDS (C-layout -> A-layout); wave-local, in-order LDS
#pragma unroll
    for (int j = 0; j < 4; ++j)
#pragma unroll
      for (int v = 0; v < 8; ++v)
        Ps[wv][v + msel * 8][j * 16 + nl] = (_Float16)p[j][v];

    // PV: accumulate O (16q x 128d), K-dim = 64 keys -> 2 steps
#pragma unroll
    for (int t = 0; t < 2; ++t) {
      v16h ap = ld_frag(&Ps[wv][nl][t * 32 + kbA], &Ps[wv][nl][t * 32 + kbA + 16]);
      v16h bfr[8];
#pragma unroll
      for (int j = 0; j < 8; ++j)
        bfr[j] = ld_frag(&Vs[j * 16 + nl][t * 32 + kbB],
                         &Vs[j * 16 + nl][t * 32 + kbB + 16]);
#pragma unroll
      for (int j = 0; j < 8; ++j) o[j] = wmma16(ap, bfr[j], o[j]);
    }
  }

  // normalize by row sum, store f32 [B,H,S,dk]
  const int q0 = qtile * 128 + wv * 16 + msel * 8;
#pragma unroll
  for (int v = 0; v < 8; ++v) {
    float inv = 1.0f / lrow[v];
#pragma unroll
    for (int j = 0; j < 8; ++j)
      O[((size_t)bh * SEQ + q0 + v) * DK + j * 16 + nl] = o[j][v] * inv;
  }
}

// ---------------------------------------------------------------------------
// Kernel 3: out = LayerNorm(O_gathered + queries) * gamma + beta
// Grid: B*S blocks, 256 threads (4 channels each).
// ---------------------------------------------------------------------------
__global__ __launch_bounds__(256)
void ln_residual_kernel(const float* __restrict__ O, const float* __restrict__ queries,
                        const float* __restrict__ gamma, const float* __restrict__ beta,
                        float* __restrict__ out) {
  const int row = blockIdx.x;
  const int b = row >> 11, s = row & (SEQ - 1);
  const int tid = threadIdx.x;
  const int d0  = tid * 4;
  const int h   = d0 >> 7, dd = d0 & (DK - 1);

  float4 ov = *(const float4*)&O[(((size_t)b * NHEAD + h) * SEQ + s) * DK + dd];
  float4 qv = *(const float4*)&queries[(size_t)row * DMODEL + d0];
  float x0 = ov.x + qv.x, x1 = ov.y + qv.y, x2 = ov.z + qv.z, x3 = ov.w + qv.w;
  float sum = x0 + x1 + x2 + x3;
  float sq  = x0 * x0 + x1 * x1 + x2 * x2 + x3 * x3;
#pragma unroll
  for (int m = 16; m >= 1; m >>= 1) {
    sum += __shfl_xor(sum, m);
    sq  += __shfl_xor(sq, m);
  }
  __shared__ float rs[8], rq[8];
  const int wv = tid >> 5, lane = tid & 31;
  if (lane == 0) { rs[wv] = sum; rq[wv] = sq; }
  __syncthreads();
  float tot = 0.f, totq = 0.f;
#pragma unroll
  for (int i = 0; i < 8; ++i) { tot += rs[i]; totq += rq[i]; }
  float mu   = tot * (1.0f / DMODEL);
  float var  = totq * (1.0f / DMODEL) - mu * mu;
  float rstd = rsqrtf(var + LN_EPS);
  float4 g  = *(const float4*)&gamma[d0];
  float4 be = *(const float4*)&beta[d0];
  float4 r;
  r.x = (x0 - mu) * rstd * g.x + be.x;
  r.y = (x1 - mu) * rstd * g.y + be.y;
  r.z = (x2 - mu) * rstd * g.z + be.z;
  r.w = (x3 - mu) * rstd * g.w + be.w;
  *(float4*)&out[(size_t)row * DMODEL + d0] = r;
}

// ---------------------------------------------------------------------------
extern "C" void kernel_launch(void* const* d_in, const int* in_sizes, int n_in,
                              void* d_out, int out_size, void* d_ws, size_t ws_size,
                              hipStream_t stream) {
  (void)in_sizes; (void)n_in; (void)out_size; (void)ws_size;
  const float* queries = (const float*)d_in[0];
  const float* keys    = (const float*)d_in[1];
  const float* values  = (const float*)d_in[2];
  const float* Wq = (const float*)d_in[3];
  const float* bq = (const float*)d_in[4];
  const float* Wk = (const float*)d_in[5];
  const float* bk = (const float*)d_in[6];
  const float* Wv = (const float*)d_in[7];
  const float* bv = (const float*)d_in[8];
  const float* gamma = (const float*)d_in[9];
  const float* beta  = (const float*)d_in[10];

  const size_t half_sz = (size_t)BATCH * NHEAD * SEQ * DK * sizeof(_Float16); // 16 MB
  char* ws = (char*)d_ws;
  _Float16* Qh  = (_Float16*)(ws);
  _Float16* Kh  = (_Float16*)(ws + half_sz);
  _Float16* Vth = (_Float16*)(ws + 2 * half_sz);
  float*    Ob  = (float*)(ws + 3 * half_sz);  // 32 MB

  dim3 g1((BATCH * SEQ) / 128, NHEAD);
  const float qscale = 0.08838834764831845f;  // 1/sqrt(128)
  qkv_proj_kernel<<<g1, 256, 0, stream>>>(queries, Wq, bq, Qh, 0, qscale);
  qkv_proj_kernel<<<g1, 256, 0, stream>>>(keys,    Wk, bk, Kh, 0, 1.0f);
  qkv_proj_kernel<<<g1, 256, 0, stream>>>(values,  Wv, bv, Vth, 1, 1.0f);
  attention_kernel<<<dim3(SEQ / 128, BATCH * NHEAD), 256, 0, stream>>>(Qh, Kh, Vth, Ob);
  ln_residual_kernel<<<dim3(BATCH * SEQ), 256, 0, stream>>>(Ob, queries, gamma, beta,
                                                            (float*)d_out);
}
